// DynamicallyQuantizedLinear_5634997092440
// MI455X (gfx1250) — 8-bit matrix operand order repaired and checked by reading, compiled, not run on hardware
//
#include <hip/hip_runtime.h>
#include <stdint.h>


typedef __attribute__((ext_vector_type(8))) int v8i;
typedef __attribute__((ext_vector_type(4))) unsigned int v4u;

#define M_TOK  4096
#define K_FEAT 4096
#define N_OUT  11008

#define NBLK_RED 1024
#define EPS_F 1.1920928955078125e-07f

#define XQ_OFF     0ull
#define XQ_BYTES   ((size_t)M_TOK * (size_t)K_FEAT)
#define WSUM_OFF   (XQ_OFF + XQ_BYTES)
#define WSUM_BYTES ((size_t)N_OUT * 4)
#define PMIN_OFF   (WSUM_OFF + WSUM_BYTES)
#define PMAX_OFF   (PMIN_OFF + (size_t)NBLK_RED * 4)
#define SZ_OFF     (PMAX_OFF + (size_t)NBLK_RED * 4)
#define WT_OFF     (((SZ_OFF + 64) + 255) & ~(size_t)255)
#define WT_BYTES   ((size_t)N_OUT * (size_t)K_FEAT)

__device__ __forceinline__ void async_b128(unsigned lds_addr,
                                           unsigned long long gaddr) {
  asm volatile("global_load_async_to_lds_b128 %0, %1, off"
               :: "v"(lds_addr), "v"(gaddr) : "memory");
}
__device__ __forceinline__ void wait_async0() {
  asm volatile("s_wait_asynccnt 0x0" ::: "memory");
}

__global__ void k_minmax(const float* __restrict__ x,
                         float* __restrict__ pmin, float* __restrict__ pmax,
                         int n4) {
  __shared__ float smin[256], smax[256];
  float lmin = 0.0f, lmax = 0.0f;
  const float4* x4 = (const float4*)x;
  for (int i = blockIdx.x * blockDim.x + threadIdx.x; i < n4;
       i += gridDim.x * blockDim.x) {
    float4 v = x4[i];
    lmin = fminf(lmin, fminf(fminf(v.x, v.y), fminf(v.z, v.w)));
    lmax = fmaxf(lmax, fmaxf(fmaxf(v.x, v.y), fmaxf(v.z, v.w)));
  }
  smin[threadIdx.x] = lmin;
  smax[threadIdx.x] = lmax;
  __syncthreads();
  for (int s = 128; s > 0; s >>= 1) {
    if ((int)threadIdx.x < s) {
      smin[threadIdx.x] = fminf(smin[threadIdx.x], smin[threadIdx.x + s]);
      smax[threadIdx.x] = fmaxf(smax[threadIdx.x], smax[threadIdx.x + s]);
    }
    __syncthreads();
  }
  if (threadIdx.x == 0) {
    pmin[blockIdx.x] = smin[0];
    pmax[blockIdx.x] = smax[0];
  }
}

__global__ void k_finalize(const float* __restrict__ pmin,
                           const float* __restrict__ pmax,
                           int nblk, float* __restrict__ sz) {
  __shared__ float smin[256], smax[256];
  float lmin = 0.0f, lmax = 0.0f;
  for (int i = threadIdx.x; i < nblk; i += 256) {
    lmin = fminf(lmin, pmin[i]);
    lmax = fmaxf(lmax, pmax[i]);
  }
  smin[threadIdx.x] = lmin;
  smax[threadIdx.x] = lmax;
  __syncthreads();
  for (int s = 128; s > 0; s >>= 1) {
    if ((int)threadIdx.x < s) {
      smin[threadIdx.x] = fminf(smin[threadIdx.x], smin[threadIdx.x + s]);
      smax[threadIdx.x] = fmaxf(smax[threadIdx.x], smax[threadIdx.x + s]);
    }
    __syncthreads();
  }
  if (threadIdx.x == 0) {
    float minv = fminf(smin[0], 0.0f);
    float maxv = fmaxf(smax[0], 0.0f);
    float scale = fmaxf((maxv - minv) / 255.0f, EPS_F);
    float zpf = rintf(-128.0f - minv / scale);
    zpf = fminf(127.0f, fmaxf(-128.0f, zpf));
    sz[0] = scale;
    sz[1] = zpf;
  }
}

__global__ void k_quant(const float* __restrict__ x,
                        const float* __restrict__ sz,
                        unsigned int* __restrict__ xq4, int n4) {
  const float inv = 1.0f / sz[0];
  const int zp = (int)sz[1];
  int i = blockIdx.x * blockDim.x + threadIdx.x;
  if (i >= n4) return;
  float4 v = ((const float4*)x)[i];
  int q0 = (int)rintf(v.x * inv) + zp;
  int q1 = (int)rintf(v.y * inv) + zp;
  int q2 = (int)rintf(v.z * inv) + zp;
  int q3 = (int)rintf(v.w * inv) + zp;
  q0 = q0 < -128 ? -128 : (q0 > 127 ? 127 : q0);
  q1 = q1 < -128 ? -128 : (q1 > 127 ? 127 : q1);
  q2 = q2 < -128 ? -128 : (q2 > 127 ? 127 : q2);
  q3 = q3 < -128 ? -128 : (q3 > 127 ? 127 : q3);
  xq4[i] = (unsigned int)(q0 & 0xff) | ((unsigned int)(q1 & 0xff) << 8) |
           ((unsigned int)(q2 & 0xff) << 16) | ((unsigned int)(q3 & 0xff) << 24);
}

__global__ void k_wsums(const int8_t* __restrict__ w, int* __restrict__ wsums) {
  int c4 = blockIdx.x * blockDim.x + threadIdx.x;
  int n = c4 * 4;
  if (n >= N_OUT) return;
  int s0 = 0, s1 = 0, s2 = 0, s3 = 0;
  const uint8_t* wp = (const uint8_t*)w + n;
  for (int k = 0; k < K_FEAT; ++k) {
    unsigned int b =
        __builtin_nontemporal_load((const unsigned int*)(wp + (size_t)k * N_OUT));
    s0 += (int)(int8_t)(b & 0xff);
    s1 += (int)(int8_t)((b >> 8) & 0xff);
    s2 += (int)(int8_t)((b >> 16) & 0xff);
    s3 += (int)(int8_t)((b >> 24) & 0xff);
  }
  wsums[n + 0] = s0;
  wsums[n + 1] = s1;
  wsums[n + 2] = s2;
  wsums[n + 3] = s3;
}

__global__ void k_transpose(const int8_t* __restrict__ w,
                            int8_t* __restrict__ wt) {
  __shared__ int8_t tile[64][68];
  const int k0 = blockIdx.x * 64;
  const int n0 = blockIdx.y * 64;
  const int r = threadIdx.x >> 2;
  const int seg = (threadIdx.x & 3) * 16;

  v4u t4 = __builtin_nontemporal_load(
      (const v4u*)(w + (size_t)(k0 + r) * N_OUT + n0 + seg));
  *(v4u*)&tile[r][seg] = t4;
  __syncthreads();

  v4u o;
#pragma unroll
  for (int j = 0; j < 4; ++j) {
    unsigned c0 = (uint8_t)tile[seg + j * 4 + 0][r];
    unsigned c1 = (uint8_t)tile[seg + j * 4 + 1][r];
    unsigned c2 = (uint8_t)tile[seg + j * 4 + 2][r];
    unsigned c3 = (uint8_t)tile[seg + j * 4 + 3][r];
    o[j] = c0 | (c1 << 8) | (c2 << 16) | (c3 << 24);
  }
  *(v4u*)(wt + (size_t)(n0 + r) * K_FEAT + k0 + seg) = o;
}

__global__ __launch_bounds__(256, 1)
void k_gemm(const int8_t* __restrict__ xq, const int8_t* __restrict__ wt,
            const int* __restrict__ wsums, const float* __restrict__ w_scales,
            const float* __restrict__ sz, float* __restrict__ y) {
  __shared__ int8_t As[2][128][64];
  __shared__ int8_t Bs[2][256][64];

  const int lane = threadIdx.x & 31;
  const int wave = threadIdx.x >> 5;
  const int wm = wave >> 2;
  const int wn = wave & 3;
  const int hi = lane >> 4;
  const int l15 = lane & 15;
  const int m0 = blockIdx.x * 128;
  const int n0 = blockIdx.y * 256;

  v8i acc[4][4];
  v8i zero = {0, 0, 0, 0, 0, 0, 0, 0};
#pragma unroll
  for (int mt = 0; mt < 4; ++mt)
#pragma unroll
    for (int nt = 0; nt < 4; ++nt) acc[mt][nt] = zero;

  const int arow = threadIdx.x >> 1;
  const int aseg = (threadIdx.x & 1) * 32;
  const int brow = threadIdx.x;
  const int8_t* ag = xq + (size_t)(m0 + arow) * K_FEAT + aseg;
  const int8_t* bg = wt + (size_t)(n0 + brow) * K_FEAT;
  unsigned lda[2], ldb[2];
  lda[0] = (unsigned)(uintptr_t)&As[0][arow][aseg];
  lda[1] = (unsigned)(uintptr_t)&As[1][arow][aseg];
  ldb[0] = (unsigned)(uintptr_t)&Bs[0][brow][0];
  ldb[1] = (unsigned)(uintptr_t)&Bs[1][brow][0];

  async_b128(lda[0],      (unsigned long long)(uintptr_t)ag);
  async_b128(lda[0] + 16, (unsigned long long)(uintptr_t)(ag + 16));
#pragma unroll
  for (int c = 0; c < 4; ++c)
    async_b128(ldb[0] + c * 16, (unsigned long long)(uintptr_t)(bg + c * 16));

  int cur = 0;
  for (int kk = 0; kk < K_FEAT; kk += 64) {
    wait_async0();
    __syncthreads();

    if (kk + 64 < K_FEAT) {
      const int nxt = cur ^ 1;
      async_b128(lda[nxt],      (unsigned long long)(uintptr_t)(ag + kk + 64));
      async_b128(lda[nxt] + 16, (unsigned long long)(uintptr_t)(ag + kk + 80));
#pragma unroll
      for (int c = 0; c < 4; ++c)
        async_b128(ldb[nxt] + c * 16,
                   (unsigned long long)(uintptr_t)(bg + kk + 64 + c * 16));
    }

    v8i bfrag[4];
#pragma unroll
    for (int nt = 0; nt < 4; ++nt) {
      int n = wn * 64 + nt * 16 + l15;
      uint2 q0 = *(const uint2*)&Bs[cur][n][8 * hi];
      uint2 q1 = *(const uint2*)&Bs[cur][n][16 + 8 * hi];
      uint2 q2 = *(const uint2*)&Bs[cur][n][32 + 8 * hi];
      uint2 q3 = *(const uint2*)&Bs[cur][n][48 + 8 * hi];
      v8i b = {(int)q0.x, (int)q0.y, (int)q1.x, (int)q1.y,
               (int)q2.x, (int)q2.y, (int)q3.x, (int)q3.y};
      bfrag[nt] = b;
    }

#pragma unroll
    for (int mt = 0; mt < 4; ++mt) {
      int row = wm * 64 + mt * 16 + l15;
      uint2 p0 = *(const uint2*)&As[cur][row][8 * hi];
      uint2 p1 = *(const uint2*)&As[cur][row][16 + 8 * hi];
      uint2 p2 = *(const uint2*)&As[cur][row][32 + 8 * hi];
      uint2 p3 = *(const uint2*)&As[cur][row][48 + 8 * hi];
      v8i a = {(int)p0.x, (int)p0.y, (int)p1.x, (int)p1.y,
               (int)p2.x, (int)p2.y, (int)p3.x, (int)p3.y};
#pragma unroll
      for (int nt = 0; nt < 4; ++nt) {
        acc[mt][nt] = __builtin_amdgcn_wmma_i32_16x16x64_iu8(
            true, a, true, bfrag[nt], acc[mt][nt], false, false);
      }
    }
    cur ^= 1;
  }

  const float xs = sz[0];
  const int zp = (int)sz[1];
#pragma unroll
  for (int nt = 0; nt < 4; ++nt) {
    int n = n0 + wn * 64 + nt * 16 + l15;
    float fs = xs * w_scales[n];
    int corr = zp * wsums[n];
#pragma unroll
    for (int mt = 0; mt < 4; ++mt) {
      int mbase = m0 + wm * 64 + mt * 16 + 8 * hi;
#pragma unroll
      for (int r = 0; r < 8; ++r) {
        __builtin_nontemporal_store(fs * (float)(acc[mt][nt][r] - corr),
                                    y + (size_t)(mbase + r) * N_OUT + n);
      }
    }
  }
}

extern "C" void kernel_launch(void* const* d_in, const int* in_sizes, int n_in,
                              void* d_out, int out_size, void* d_ws,
                              size_t ws_size, hipStream_t stream) {
  (void)in_sizes; (void)n_in; (void)out_size; (void)ws_size;
  const float*  x   = (const float*)d_in[0];
  const int8_t* w   = (const int8_t*)d_in[1];
  const float*  wsc = (const float*)d_in[2];
  float* y = (float*)d_out;

  char* ws = (char*)d_ws;
  int8_t* xq   = (int8_t*)(ws + XQ_OFF);
  int*   wsums = (int*)(ws + WSUM_OFF);
  float* pmin  = (float*)(ws + PMIN_OFF);
  float* pmax  = (float*)(ws + PMAX_OFF);
  float* sz    = (float*)(ws + SZ_OFF);
  int8_t* wt   = (int8_t*)(ws + WT_OFF);

  const int n_elem = M_TOK * K_FEAT;
  const int n4 = n_elem / 4;

  k_minmax<<<NBLK_RED, 256, 0, stream>>>(x, pmin, pmax, n4);
  k_finalize<<<1, 256, 0, stream>>>(pmin, pmax, NBLK_RED, sz);
  k_quant<<<(n4 + 255) / 256, 256, 0, stream>>>(x, sz, (unsigned int*)xq, n4);
  k_wsums<<<(N_OUT / 4 + 255) / 256, 256, 0, stream>>>(w, wsums);

  dim3 tgrid(K_FEAT / 64, N_OUT / 64);
  k_transpose<<<tgrid, 256, 0, stream>>>(w, wt);

  dim3 grid(M_TOK / 128, N_OUT / 256);
  k_gemm<<<grid, 256, 0, stream>>>(xq, wt, wsums, wsc, sz, y);
}
